// WindowAttentionGlobal_86904368267308
// MI455X (gfx1250) — matrix operand order repaired and checked by reading, not compiled, not run on hardware
//
#include <hip/hip_runtime.h>

// ---------------------------------------------------------------------------
// Types for CDNA5 WMMA (gfx1250, wave32)
// ---------------------------------------------------------------------------
typedef __bf16 bf16;
typedef float  v8f   __attribute__((ext_vector_type(8)));
typedef __bf16 v16bf __attribute__((ext_vector_type(16)));
typedef unsigned int u32x4 __attribute__((ext_vector_type(4)));

union ABReg { v16bf v; u32x4 q[2]; };

// A-operand (16x32 bf16, MxK): lane<16 holds row=lane, K=0..15 contiguous;
// lane>=16 holds row=lane-16, K=16..31. Two 16B chunks per lane.
__device__ __forceinline__ v16bf load_A_16x32(const bf16* base, int ld) {
    const int lane = threadIdx.x & 31;
    const int row  = lane & 15;
    const int k0   = (lane & 16) ? 16 : 0;
    const bf16* p  = base + (size_t)row * ld + k0;
    ABReg r;
    r.q[0] = *reinterpret_cast<const u32x4*>(p);
    r.q[1] = *reinterpret_cast<const u32x4*>(p + 8);
    return r.v;
}

// B-operand (32x16 bf16, KxN): lane<16 holds col=lane, K=0..15 contiguous;
// lane>=16 holds col=lane-16, K=16..31. Memory layout: [n][k], k contiguous.
__device__ __forceinline__ v16bf load_B_32x16(const bf16* base, int ld) {
    const int lane = threadIdx.x & 31;
    const int col  = lane & 15;
    const int k0   = (lane & 16) ? 16 : 0;
    const bf16* p  = base + (size_t)col * ld + k0;
    ABReg r;
    r.q[0] = *reinterpret_cast<const u32x4*>(p);
    r.q[1] = *reinterpret_cast<const u32x4*>(p + 8);
    return r.v;
}

__device__ __forceinline__ v8f wmma_bf16(v16bf a, v16bf b, v8f c) {
    return __builtin_amdgcn_wmma_f32_16x16x32_bf16(
        /*neg_a=*/false, a, /*neg_b=*/false, b,
        /*c_mod=*/(short)0, c, /*reuse_a=*/false, /*reuse_b=*/false);
}

// ---------------------------------------------------------------------------
// Problem constants (fixed by setup_inputs)
// ---------------------------------------------------------------------------
#define BATCH     8
#define BW        16
#define BTOT      128     // B_ = 8*16
#define HEADS     16
#define HDIM      32
#define NTOK      256     // wh*ww = 8*32
#define CDIM      512
#define MROWS     32768   // BTOT*NTOK
#define QSCALE    0.17677669529663687f   // 32^-0.5

// ---------------------------------------------------------------------------
// Prep: f32 -> bf16 conversions (optionally scaled)
// ---------------------------------------------------------------------------
__global__ __launch_bounds__(256) void cvt_bf16_kernel(const float* __restrict__ in,
                                                       bf16* __restrict__ out,
                                                       int n, float scale) {
    int i = blockIdx.x * 256 + threadIdx.x;
    if (i < n) out[i] = (bf16)(in[i] * scale);
}

// bias[h][n][m] = rpb_table[(nh-mh+7)*63 + (nw-mw+31)][h]
__global__ __launch_bounds__(256) void bias_build_kernel(const float* __restrict__ rpb,
                                                         float* __restrict__ bias) {
    int idx = blockIdx.x * 256 + threadIdx.x;     // over HEADS*NTOK*NTOK = 1M
    if (idx >= HEADS * NTOK * NTOK) return;
    int h  = idx >> 16;
    int nm = idx & 65535;
    int n  = nm >> 8, m = nm & 255;
    int nh = n >> 5, nw = n & 31;
    int mh = m >> 5, mw = m & 31;
    int r  = (nh - mh + 7) * 63 + (nw - mw + 31);
    bias[idx] = rpb[r * HEADS + h];
}

// ---------------------------------------------------------------------------
// QKV GEMM: kv[t,c] = sum_i x[t,i] * qkv_w[c,i] + qkv_b[c]
// Writes K  -> [B_,H,N,hd] bf16   (B-layout source for QK^T)
//        V^T-> [B_,H,hd,N] bf16   (B-layout source for AV)
// Block: 256 threads (8 waves); block tile 64 rows x 128 cols; wave 32x32.
// ---------------------------------------------------------------------------
__global__ __launch_bounds__(256) void qkv_gemm_kernel(
    const bf16* __restrict__ xb, const bf16* __restrict__ wb,
    const float* __restrict__ qkv_b,
    bf16* __restrict__ k_ws, bf16* __restrict__ vT_ws) {
    const int lane = threadIdx.x & 31;
    const int wave = threadIdx.x >> 5;
    const int row0 = blockIdx.x * 64 + (wave >> 2) * 32;
    const int col0 = blockIdx.y * 128 + (wave & 3) * 32;

    v8f acc[2][2] = {};
    for (int kk = 0; kk < CDIM; kk += 32) {
        if (kk + 32 < CDIM) {  // gfx1250 global_prefetch_b8
            __builtin_prefetch(xb + (size_t)row0 * CDIM + kk + 32, 0, 1);
        }
        v16bf a0 = load_A_16x32(xb + (size_t)row0 * CDIM + kk, CDIM);
        v16bf a1 = load_A_16x32(xb + (size_t)(row0 + 16) * CDIM + kk, CDIM);
        v16bf b0 = load_B_32x16(wb + (size_t)col0 * CDIM + kk, CDIM);
        v16bf b1 = load_B_32x16(wb + (size_t)(col0 + 16) * CDIM + kk, CDIM);
        acc[0][0] = wmma_bf16(a0, b0, acc[0][0]);
        acc[0][1] = wmma_bf16(a0, b1, acc[0][1]);
        acc[1][0] = wmma_bf16(a1, b0, acc[1][0]);
        acc[1][1] = wmma_bf16(a1, b1, acc[1][1]);
    }

    const int colLane = lane & 15;
    const int rbase   = (lane & 16) ? 8 : 0;
#pragma unroll
    for (int i = 0; i < 2; ++i)
#pragma unroll
        for (int j = 0; j < 2; ++j) {
#pragma unroll
            for (int r = 0; r < 8; ++r) {
                int t = row0 + i * 16 + rbase + r;         // token row
                int c = col0 + j * 16 + colLane;           // 0..1023
                float val = acc[i][j][r] + qkv_b[c];
                int b_ = t >> 8, n = t & 255;
                if (c < CDIM) {                            // K
                    int h = c >> 5, d = c & 31;
                    k_ws[(((size_t)b_ * HEADS + h) * NTOK + n) * HDIM + d] = (bf16)val;
                } else {                                   // V (transposed)
                    int cc = c - CDIM, h = cc >> 5, d = cc & 31;
                    vT_ws[(((size_t)b_ * HEADS + h) * HDIM + d) * NTOK + n] = (bf16)val;
                }
            }
        }
}

// ---------------------------------------------------------------------------
// Fused attention: per block = one (b_, h). K, V^T staged in LDS.
// Each of 8 waves owns 32 query rows (2 row-tiles of 16).
// attn = q*scale @ k^T + bias (bias loaded as WMMA C), softmax, @ v.
// ---------------------------------------------------------------------------
__global__ __launch_bounds__(256) void attn_kernel(
    const bf16* __restrict__ qb,     // [8][16][256][32] pre-scaled
    const bf16* __restrict__ k_ws,   // [128][16][256][32]
    const bf16* __restrict__ vT_ws,  // [128][16][32][256]
    const float* __restrict__ bias,  // [16][256][256]
    bf16* __restrict__ o_ws) {       // [128][256][512]  (=[b_][n][h*32+d])
    const int bh = blockIdx.x;       // 0..2047
    const int b_ = bh >> 4, h = bh & 15;
    const int b  = b_ >> 4;          // image index (BW=16)

    __shared__ alignas(32) bf16 kbuf[NTOK * HDIM];   // [m][d], 16 KB
    __shared__ alignas(32) bf16 vbuf[HDIM * NTOK];   // [d][m], 16 KB
    __shared__ alignas(32) bf16 pbuf[8][16 * NTOK];  // per-wave probs, 64 KB

    // Cooperative stage of K and V^T (1024 x 16B each)
    {
        const bf16* kg = k_ws + (size_t)bh * NTOK * HDIM;
        const bf16* vg = vT_ws + (size_t)bh * HDIM * NTOK;
        for (int i = threadIdx.x; i < (NTOK * HDIM) / 8; i += 256) {
            reinterpret_cast<u32x4*>(kbuf)[i] = reinterpret_cast<const u32x4*>(kg)[i];
            reinterpret_cast<u32x4*>(vbuf)[i] = reinterpret_cast<const u32x4*>(vg)[i];
        }
    }
    __syncthreads();

    const int lane    = threadIdx.x & 31;
    const int wave    = threadIdx.x >> 5;
    const int colLane = lane & 15;
    const int rbase   = (lane & 16) ? 8 : 0;
    const float* bias_h = bias + (size_t)h * NTOK * NTOK;

    for (int it = 0; it < 2; ++it) {
        const int row0 = wave * 32 + it * 16;   // query row tile base

        // Q tile: 16 rows x full hd=32 — one WMMA K-step
        v16bf aq = load_A_16x32(
            qb + ((((size_t)b * HEADS + h) * NTOK) + row0) * HDIM, HDIM);

        // ---- S = q k^T + bias : 16 tiles of 16x16 across m ----
        v8f s[16];
#pragma unroll
        for (int mt = 0; mt < 16; ++mt) {
            v8f c;
#pragma unroll
            for (int r = 0; r < 8; ++r)
                c[r] = bias_h[(size_t)(row0 + rbase + r) * NTOK + mt * 16 + colLane];
            v16bf bk = load_B_32x16(kbuf + (mt * 16) * HDIM, HDIM);
            s[mt] = wmma_bf16(aq, bk, c);
        }

        // ---- softmax over m (rows live in 16-lane half-wave groups) ----
        float mx[8], sm[8];
#pragma unroll
        for (int r = 0; r < 8; ++r) {
            float m = -3.0e38f;
#pragma unroll
            for (int mt = 0; mt < 16; ++mt) m = fmaxf(m, s[mt][r]);
#pragma unroll
            for (int d = 1; d < 16; d <<= 1) m = fmaxf(m, __shfl_xor(m, d, 32));
            mx[r] = m;
        }
#pragma unroll
        for (int r = 0; r < 8; ++r) {
            float t = 0.f;
#pragma unroll
            for (int mt = 0; mt < 16; ++mt) {
                float e = __expf(s[mt][r] - mx[r]);
                s[mt][r] = e;
                t += e;
            }
#pragma unroll
            for (int d = 1; d < 16; d <<= 1) t += __shfl_xor(t, d, 32);
            sm[r] = 1.0f / t;
        }

        // ---- write probs to LDS (D-layout -> row-major), bf16 ----
        bf16* pw = pbuf[wave];
#pragma unroll
        for (int mt = 0; mt < 16; ++mt)
#pragma unroll
            for (int r = 0; r < 8; ++r)
                pw[(rbase + r) * NTOK + mt * 16 + colLane] = (bf16)(s[mt][r] * sm[r]);

        // ---- O = P @ V : A from pbuf (transposed read), B from vbuf ----
        v8f o0 = {}, o1 = {};
#pragma unroll
        for (int mc = 0; mc < 8; ++mc) {   // m in chunks of 32
            v16bf ap = load_A_16x32(pw + mc * 32, NTOK);
            o0 = wmma_bf16(ap, load_B_32x16(vbuf + mc * 32, NTOK), o0);
            o1 = wmma_bf16(ap, load_B_32x16(vbuf + 16 * NTOK + mc * 32, NTOK), o1);
        }

        // ---- store bf16 output in [b_][n][h*32+d] for the proj GEMM ----
        bf16* og = o_ws + (size_t)b_ * NTOK * CDIM + (size_t)h * HDIM;
#pragma unroll
        for (int r = 0; r < 8; ++r) {
            size_t nrow = (size_t)(row0 + rbase + r) * CDIM;
            og[nrow + colLane]      = (bf16)o0[r];
            og[nrow + 16 + colLane] = (bf16)o1[r];
        }
    }
}

// ---------------------------------------------------------------------------
// Output projection: out[t,c] = sum_i o[t,i] * proj_w[c,i] + proj_b[c]  (f32)
// ---------------------------------------------------------------------------
__global__ __launch_bounds__(256) void proj_gemm_kernel(
    const bf16* __restrict__ ob, const bf16* __restrict__ wb,
    const float* __restrict__ proj_b, float* __restrict__ out) {
    const int lane = threadIdx.x & 31;
    const int wave = threadIdx.x >> 5;
    const int row0 = blockIdx.x * 64 + (wave >> 2) * 32;
    const int col0 = blockIdx.y * 128 + (wave & 3) * 32;

    v8f acc[2][2] = {};
    for (int kk = 0; kk < CDIM; kk += 32) {
        if (kk + 32 < CDIM) {
            __builtin_prefetch(ob + (size_t)row0 * CDIM + kk + 32, 0, 1);
        }
        v16bf a0 = load_A_16x32(ob + (size_t)row0 * CDIM + kk, CDIM);
        v16bf a1 = load_A_16x32(ob + (size_t)(row0 + 16) * CDIM + kk, CDIM);
        v16bf b0 = load_B_32x16(wb + (size_t)col0 * CDIM + kk, CDIM);
        v16bf b1 = load_B_32x16(wb + (size_t)(col0 + 16) * CDIM + kk, CDIM);
        acc[0][0] = wmma_bf16(a0, b0, acc[0][0]);
        acc[0][1] = wmma_bf16(a0, b1, acc[0][1]);
        acc[1][0] = wmma_bf16(a1, b0, acc[1][0]);
        acc[1][1] = wmma_bf16(a1, b1, acc[1][1]);
    }

    const int colLane = lane & 15;
    const int rbase   = (lane & 16) ? 8 : 0;
#pragma unroll
    for (int i = 0; i < 2; ++i)
#pragma unroll
        for (int j = 0; j < 2; ++j)
#pragma unroll
            for (int r = 0; r < 8; ++r) {
                int t = row0 + i * 16 + rbase + r;
                int c = col0 + j * 16 + colLane;
                out[(size_t)t * CDIM + c] = acc[i][j][r] + proj_b[c];
            }
}

// ---------------------------------------------------------------------------
// Host launcher
// ---------------------------------------------------------------------------
extern "C" void kernel_launch(void* const* d_in, const int* in_sizes, int n_in,
                              void* d_out, int out_size, void* d_ws, size_t ws_size,
                              hipStream_t stream) {
    const float* x      = (const float*)d_in[0];   // [128,256,512]
    const float* qg     = (const float*)d_in[1];   // [8,1,16,256,32]
    const float* qkv_w  = (const float*)d_in[2];   // [1024,512]
    const float* qkv_b  = (const float*)d_in[3];   // [1024]
    const float* proj_w = (const float*)d_in[4];   // [512,512]
    const float* proj_b = (const float*)d_in[5];   // [512]
    const float* rpb    = (const float*)d_in[6];   // [945,16]
    float* out = (float*)d_out;

    char* ws = (char*)d_ws;
    size_t off = 0;
    auto take = [&](size_t bytes) -> char* {
        char* p = ws + off;
        off = (off + bytes + 255) & ~(size_t)255;
        return p;
    };
    bf16* xb      = (bf16*)take((size_t)MROWS * CDIM * 2);        // 32 MB
    bf16* qkvw_b  = (bf16*)take((size_t)2 * CDIM * CDIM * 2);     // 1 MB
    bf16* projw_b = (bf16*)take((size_t)CDIM * CDIM * 2);         // 0.5 MB
    bf16* qb      = (bf16*)take((size_t)BATCH * HEADS * NTOK * HDIM * 2);  // 2 MB
    float* biasw  = (float*)take((size_t)HEADS * NTOK * NTOK * 4);         // 4 MB
    bf16* k_ws    = (bf16*)take((size_t)BTOT * HEADS * NTOK * HDIM * 2);   // 32 MB
    bf16* vT_ws   = (bf16*)take((size_t)BTOT * HEADS * HDIM * NTOK * 2);   // 32 MB
    bf16* o_ws    = (bf16*)take((size_t)MROWS * CDIM * 2);                 // 32 MB

    // ---- prep ----
    {
        int n = MROWS * CDIM;              // 16,777,216
        cvt_bf16_kernel<<<(n + 255) / 256, 256, 0, stream>>>(x, xb, n, 1.0f);
    }
    {
        int n = 2 * CDIM * CDIM;           // 524,288
        cvt_bf16_kernel<<<(n + 255) / 256, 256, 0, stream>>>(qkv_w, qkvw_b, n, 1.0f);
    }
    {
        int n = CDIM * CDIM;               // 262,144
        cvt_bf16_kernel<<<(n + 255) / 256, 256, 0, stream>>>(proj_w, projw_b, n, 1.0f);
    }
    {
        int n = BATCH * HEADS * NTOK * HDIM;  // 1,048,576 (pre-scale q)
        cvt_bf16_kernel<<<(n + 255) / 256, 256, 0, stream>>>(qg, qb, n, QSCALE);
    }
    {
        int n = HEADS * NTOK * NTOK;       // 1,048,576
        bias_build_kernel<<<(n + 255) / 256, 256, 0, stream>>>(rpb, biasw);
    }

    // ---- QKV projection ----
    {
        dim3 grid(MROWS / 64, (2 * CDIM) / 128);   // 512 x 8
        qkv_gemm_kernel<<<grid, 256, 0, stream>>>(xb, qkvw_b, qkv_b, k_ws, vT_ws);
    }

    // ---- fused attention ----
    {
        dim3 grid(BTOT * HEADS);                    // 2048
        attn_kernel<<<grid, 256, 0, stream>>>(qb, k_ws, vT_ws, biasw, o_ws);
    }

    // ---- output projection ----
    {
        dim3 grid(MROWS / 64, CDIM / 128);          // 512 x 4
        proj_gemm_kernel<<<grid, 256, 0, stream>>>(o_ws, projw_b, proj_b, out);
    }
}
